// PointNetMSGDown3d_14482629722278
// MI455X (gfx1250) — matrix operand order repaired and checked by reading, not compiled, not run on hardware
//
#include <hip/hip_runtime.h>

#define B_      4
#define T_      2
#define N_      8192
#define NPOINT_ 2048
#define M_      (T_ * NPOINT_)   // 4096
#define KMAX_   64

typedef __attribute__((ext_vector_type(16))) _Float16 v16h;
typedef __attribute__((ext_vector_type(2)))  _Float16 h2;
typedef __attribute__((ext_vector_type(8)))  float    v8f;
typedef __attribute__((ext_vector_type(8)))  unsigned v8u;

__device__ __forceinline__ v8f wmma_f32_16x16x32_f16(v16h a, v16h b, v8f c) {
  return __builtin_amdgcn_wmma_f32_16x16x32_f16(false, a, false, b, (short)0, c, false, false);
}

__device__ __forceinline__ float lrelu(float x) { return x >= 0.f ? x : 0.01f * x; }

// ---- LDS-free cross-lane helpers (wave32) --------------------------------
__device__ __forceinline__ unsigned permx16_u(unsigned v) {   // lane ^= 16
  return __builtin_amdgcn_permlanex16(v, v, 0x76543210u, 0xFEDCBA98u, false, false);
}
template <int MASK>
__device__ __forceinline__ unsigned permxor_u(unsigned v) {   // lane ^= MASK
  if constexpr (MASK == 1)
    return __builtin_amdgcn_permlane16(v, v, 0x67452301u, 0xEFCDAB89u, false, false);
  else if constexpr (MASK == 2)
    return __builtin_amdgcn_permlane16(v, v, 0x54761032u, 0xDCFE98BAu, false, false);
  else if constexpr (MASK == 4)
    return __builtin_amdgcn_permlane16(v, v, 0x32107654u, 0xBA98FEDCu, false, false);
  else if constexpr (MASK == 8)
    return __builtin_amdgcn_permlane16(v, v, 0xFEDCBA98u, 0x76543210u, false, false);
  else
    return permx16_u(v);
}
template <int MASK>
__device__ __forceinline__ float permxor_f(float x) {
  return __uint_as_float(permxor_u<MASK>(__float_as_uint(x)));
}
template <int MASK>
__device__ __forceinline__ int permxor_i(int x) {
  return (int)permxor_u<MASK>((unsigned)x);
}

__device__ __forceinline__ unsigned packh2(float a, float b) {  // v_cvt_pk_f16_f32
  h2 p; p.x = (_Float16)a; p.y = (_Float16)b;
  return __builtin_bit_cast(unsigned, p);
}

// C-layout (two 16-row tiles) -> B-layout (32x16 f16) via packed xor-16 swap.
__device__ __forceinline__ v16h make_bfrag(const float* a0, const float* a1, int hi) {
  v8u bu;
#pragma unroll
  for (int v2 = 0; v2 < 4; v2++) {
    unsigned p0 = packh2(a0[2 * v2], a0[2 * v2 + 1]);
    unsigned p1 = packh2(a1[2 * v2], a1[2 * v2 + 1]);
    unsigned s0 = permx16_u(p0);
    unsigned s1 = permx16_u(p1);
    bu[v2]     = hi ? s1 : p0;
    bu[v2 + 4] = hi ? p1 : s0;
  }
  return __builtin_bit_cast(v16h, bu);
}

// ---------------------------------------------------------------------------
// Kernel 1: furthest point sampling. One 1024-thread block per cloud (B*T=8).
// dists lives in LDS (32KB). 2047 sequential steps of min-update + argmax.
// Argmax butterfly uses v_permlane16/x16 (no LDS traffic in the hot loop).
// ---------------------------------------------------------------------------
__global__ __launch_bounds__(1024)
void fps_kernel(const float* __restrict__ xyz, int* __restrict__ ind) {
  const int cloud = blockIdx.x;
  const float* px = xyz + (size_t)cloud * 3 * N_;
  __shared__ float dists[N_];
  __shared__ float rv[32];
  __shared__ int   ri[32];
  __shared__ int   lastS;
  const int tid = threadIdx.x;
  const int lane = tid & 31, wv = tid >> 5;

  for (int n = tid; n < N_; n += 1024) dists[n] = 1.0e10f;
  if (tid == 0) { lastS = 0; ind[(size_t)cloud * NPOINT_] = 0; }
  __syncthreads();

  int last = 0;
  for (int it = 1; it < NPOINT_; ++it) {
    const float lx = px[last], ly = px[N_ + last], lz = px[2 * N_ + last];
    float bv = -1.f; int bi = N_;
    for (int n = tid; n < N_; n += 1024) {
      float dx = px[n] - lx, dy = px[N_ + n] - ly, dz = px[2 * N_ + n] - lz;
      float d = dx * dx + dy * dy + dz * dz;
      float nd = fminf(dists[n], d);
      dists[n] = nd;
      if (nd > bv || (nd == bv && n < bi)) { bv = nd; bi = n; }
    }
#define FPS_RED(MASK)                                             \
    { float ov = permxor_f<MASK>(bv); int oi = permxor_i<MASK>(bi); \
      if (ov > bv || (ov == bv && oi < bi)) { bv = ov; bi = oi; } }
    FPS_RED(16) FPS_RED(8) FPS_RED(4) FPS_RED(2) FPS_RED(1)
    if (lane == 0) { rv[wv] = bv; ri[wv] = bi; }
    __syncthreads();
    if (wv == 0) {
      bv = rv[lane]; bi = ri[lane];
      FPS_RED(16) FPS_RED(8) FPS_RED(4) FPS_RED(2) FPS_RED(1)
      if (lane == 0) { ind[(size_t)cloud * NPOINT_ + it] = bi; lastS = bi; }
    }
#undef FPS_RED
    __syncthreads();
    last = lastS;
  }
}

// ---------------------------------------------------------------------------
// Kernel 2: gather sampled coords/times into d_out; build points + |p|^2 in ws.
// ---------------------------------------------------------------------------
__global__ __launch_bounds__(256)
void gather_kernel(const float* __restrict__ xyz, const float* __restrict__ times,
                   const int* __restrict__ ind, float* __restrict__ out,
                   float* __restrict__ pointsw, float* __restrict__ pnormw) {
  int i = blockIdx.x * 256 + threadIdx.x;       // over B*T*NPOINT
  if (i >= B_ * T_ * NPOINT_) return;
  int s = i % NPOINT_;
  int bt = i / NPOINT_;
  int t = bt % T_, b = bt / T_;
  int idx = ind[(size_t)bt * NPOINT_ + s];
  float x = xyz[((size_t)bt * 3 + 0) * N_ + idx];
  float y = xyz[((size_t)bt * 3 + 1) * N_ + idx];
  float z = xyz[((size_t)bt * 3 + 2) * N_ + idx];
  out[((size_t)bt * 3 + 0) * NPOINT_ + s] = x;
  out[((size_t)bt * 3 + 1) * NPOINT_ + s] = y;
  out[((size_t)bt * 3 + 2) * NPOINT_ + s] = z;
  out[(size_t)B_ * T_ * 3 * NPOINT_ + (size_t)bt * NPOINT_ + s] =
      times[(size_t)bt * N_ + idx];
  int m = t * NPOINT_ + s;
  pointsw[((size_t)b * 3 + 0) * M_ + m] = x;
  pointsw[((size_t)b * 3 + 1) * M_ + m] = y;
  pointsw[((size_t)b * 3 + 2) * M_ + m] = z;
  pnormw[(size_t)b * M_ + m] = x * x + y * y + z * z;
}

// ---------------------------------------------------------------------------
// Kernel 3: fold bias/affine into per-channel alpha/beta.
// ---------------------------------------------------------------------------
__global__ void ab_kernel(const float* b1, const float* g1, const float* be1,
                          const float* b2, const float* g2, const float* be2,
                          const float* b3, const float* g3, const float* be3,
                          float* alpha, float* beta) {
  int i = threadIdx.x;
  if (i >= 256) return;
  int br = i >> 7, c = i & 127;
  float g, bb, be;
  if (c < 32)       { g = g1[br * 32 + c];        bb = b1[br * 32 + c];        be = be1[br * 32 + c]; }
  else if (c < 64)  { int c2 = c - 32; g = g2[br * 32 + c2]; bb = b2[br * 32 + c2]; be = be2[br * 32 + c2]; }
  else              { int c3 = c - 64; g = g3[br * 64 + c3]; bb = b3[br * 64 + c3]; be = be3[br * 64 + c3]; }
  float al = g * rsqrtf(1.0f + 0.001f);
  alpha[i] = al;
  beta[i]  = al * bb + be;
}

// ---------------------------------------------------------------------------
// Kernel 4: exact 64-NN per query row. One wave per row; sorted 64-entry list
// of packed (orderable-dist-bits<<32 | idx) keys in LDS; ballot-driven
// cooperative insertion (broadcast via v_readlane, not LDS).
// ---------------------------------------------------------------------------
__global__ __launch_bounds__(256)
void knn_kernel(const float* __restrict__ xyz, const float* __restrict__ pointsw,
                const float* __restrict__ pnormw, int* __restrict__ nindf, int frame) {
  __shared__ unsigned long long lst[8 * 64];
  const int lane = threadIdx.x & 31, w = threadIdx.x >> 5;
  unsigned long long* L = lst + w * 64;
  const int rows = B_ * M_;
  for (int row = blockIdx.x * 8 + w; row < rows; row += gridDim.x * 8) {
    const int b = row / M_, m = row % M_;
    const float pxv = pointsw[((size_t)b * 3 + 0) * M_ + m];
    const float pyv = pointsw[((size_t)b * 3 + 1) * M_ + m];
    const float pzv = pointsw[((size_t)b * 3 + 2) * M_ + m];
    const float pn  = pnormw[(size_t)b * M_ + m];
    const float* sup = xyz + ((size_t)(b * T_ + frame) * 3) * N_;
    L[2 * lane] = ~0ull;
    L[2 * lane + 1] = ~0ull;
    for (int c0 = 0; c0 < N_; c0 += 32) {
      int n = c0 + lane;
      int pf = (c0 + 1056 < N_) ? n + 1024 : n;
      __builtin_prefetch(&sup[pf], 0, 0);
      __builtin_prefetch(&sup[N_ + pf], 0, 0);
      __builtin_prefetch(&sup[2 * N_ + pf], 0, 0);
      float sx = sup[n], sy = sup[N_ + n], sz = sup[2 * N_ + n];
      float d = pn + (sx * sx + sy * sy + sz * sz)
                   - 2.f * (pxv * sx + pyv * sy + pzv * sz);
      unsigned u = __float_as_uint(d);
      u = (u & 0x80000000u) ? ~u : (u | 0x80000000u);   // orderable bits
      unsigned long long key = ((unsigned long long)u << 32) | (unsigned)n;
      unsigned long long thr = L[63];
      unsigned long long msk = __ballot(key < thr);
      while (msk) {
        int srcl = __ffsll((long long)msk) - 1;
        unsigned klo = (unsigned)__builtin_amdgcn_readlane((int)(unsigned)(key & 0xffffffffull), srcl);
        unsigned khi = (unsigned)__builtin_amdgcn_readlane((int)(unsigned)(key >> 32), srcl);
        unsigned long long kk = ((unsigned long long)khi << 32) | klo;
        unsigned long long e0 = L[2 * lane], e1 = L[2 * lane + 1];
        unsigned long long em1 = lane ? L[2 * lane - 1] : 0ull;
        int pos = __popcll(__ballot(e0 < kk)) + __popcll(__ballot(e1 < kk));
        unsigned long long n0 = (2 * lane < pos) ? e0 : ((2 * lane == pos) ? kk : em1);
        unsigned long long n1 = (2 * lane + 1 < pos) ? e1 : ((2 * lane + 1 == pos) ? kk : e0);
        L[2 * lane] = n0;
        L[2 * lane + 1] = n1;
        msk &= ~(1ull << srcl);
        thr = L[63];
        msk &= __ballot(key < thr);
      }
    }
    int base = row * 64;
    nindf[base + 2 * lane]     = (int)(unsigned)(L[2 * lane] & 0xffffffffull);
    nindf[base + 2 * lane + 1] = (int)(unsigned)(L[2 * lane + 1] & 0xffffffffull);
  }
}

// ---------------------------------------------------------------------------
// Kernel 5: gather + 3-layer MLP (WMMA f32_16x16x32_f16) + max over k.
// One wave per query position; 16 (position,sample) columns per WMMA chunk.
// Layer transitions are pure VALU: v_cvt_pk_f16_f32 + v_permlanex16 + cndmask.
// frame 0 writes, frame 1 subtracts (frames[0]-frames[1]).
// ---------------------------------------------------------------------------
__global__ __launch_bounds__(256)
void mlp_kernel(const float* __restrict__ xyz, const float* __restrict__ feat,
                const float* __restrict__ pointsw, const int* __restrict__ nindf,
                const float* __restrict__ W1, const float* __restrict__ W2,
                const float* __restrict__ W3, const float* __restrict__ alpha,
                const float* __restrict__ beta, float* __restrict__ outfeat,
                int frame, int branch, int ksamp) {
  const int lane = threadIdx.x & 31;
  const int widx = threadIdx.x >> 5;
  const int hi   = lane >> 4;      // which 16-lane half
  const int col  = lane & 15;      // column within tile

  // --- weight fragments (A-matrix 16x32 f16 layout) ---
  v16h a1[2], a2[2], a3[4];
  {
    const float* w1 = W1 + (size_t)branch * 32 * 4;
    const float* w2 = W2 + (size_t)branch * 32 * 32;
    const float* w3 = W3 + (size_t)branch * 64 * 32;
#pragma unroll
    for (int t = 0; t < 2; t++) {
#pragma unroll
      for (int h = 0; h < 16; h++) {
        int K = h + hi * 16;
        int row = t * 16 + col;
        a1[t][h] = (_Float16)((K < 4) ? w1[row * 4 + K] : 0.f);
        a2[t][h] = (_Float16)w2[row * 32 + K];
      }
    }
#pragma unroll
    for (int t = 0; t < 4; t++)
#pragma unroll
      for (int h = 0; h < 16; h++) {
        int K = h + hi * 16;
        int row = t * 16 + col;
        a3[t][h] = (_Float16)w3[row * 32 + K];
      }
  }
  const float* ab_a = alpha + branch * 128;
  const float* ab_b = beta + branch * 128;

  const int gwave = blockIdx.x * 8 + widx;
  const int nwaves = gridDim.x * 8;
  for (int pos = gwave; pos < B_ * M_; pos += nwaves) {
    const int b = pos / M_, m = pos % M_;
    const float pxv = pointsw[((size_t)b * 3 + 0) * M_ + m];
    const float pyv = pointsw[((size_t)b * 3 + 1) * M_ + m];
    const float pzv = pointsw[((size_t)b * 3 + 2) * M_ + m];
    const float* sup = xyz + ((size_t)(b * T_ + frame) * 3) * N_;
    const float* ft  = feat + (size_t)(b * T_ + frame) * N_;
    const int* nb = nindf + (size_t)pos * KMAX_;

    float rm[4][8];
#pragma unroll
    for (int t = 0; t < 4; t++)
#pragma unroll
      for (int r = 0; r < 8; r++) rm[t][r] = -3.0e38f;

    v8f cz;
#pragma unroll
    for (int r = 0; r < 8; r++) cz[r] = 0.f;

    const int nch = ksamp >> 4;
    for (int cchunk = 0; cchunk < nch; ++cchunk) {
      // ---- gather fg column (B-matrix 32x16 f16 layout; K>=4 zero) ----
      int s = cchunk * 16 + col;
      int ni = nb[s];
      float f0 = sup[ni] - pxv;
      float f1 = sup[N_ + ni] - pyv;
      float f2 = sup[2 * N_ + ni] - pzv;
      float f3 = ft[ni];
      v8u bfu;
#pragma unroll
      for (int v2 = 0; v2 < 8; v2++) bfu[v2] = 0u;
      bfu[0] = hi ? 0u : packh2(f0, f1);
      bfu[1] = hi ? 0u : packh2(f2, f3);
      v16h bf = __builtin_bit_cast(v16h, bfu);

      // ---- layer 1 (4 -> 32) ----
      v8f x0 = wmma_f32_16x16x32_f16(a1[0], bf, cz);
      v8f x1 = wmma_f32_16x16x32_f16(a1[1], bf, cz);
      float act0[8], act1[8];
#pragma unroll
      for (int r = 0; r < 8; r++) {
        int o0 = r + hi * 8, o1 = 16 + r + hi * 8;
        act0[r] = lrelu(ab_a[o0] * x0[r] + ab_b[o0]);
        act1[r] = lrelu(ab_a[o1] * x1[r] + ab_b[o1]);
      }
      v16h b2 = make_bfrag(act0, act1, hi);

      // ---- layer 2 (32 -> 32) ----
      v8f y0 = wmma_f32_16x16x32_f16(a2[0], b2, cz);
      v8f y1 = wmma_f32_16x16x32_f16(a2[1], b2, cz);
#pragma unroll
      for (int r = 0; r < 8; r++) {
        int o0 = 32 + r + hi * 8, o1 = 48 + r + hi * 8;
        act0[r] = lrelu(ab_a[o0] * y0[r] + ab_b[o0]);
        act1[r] = lrelu(ab_a[o1] * y1[r] + ab_b[o1]);
      }
      v16h b3 = make_bfrag(act0, act1, hi);

      // ---- layer 3 (32 -> 64) + running max over samples ----
#pragma unroll
      for (int t = 0; t < 4; t++) {
        v8f zt = wmma_f32_16x16x32_f16(a3[t], b3, cz);
#pragma unroll
        for (int r = 0; r < 8; r++) {
          int o = 64 + t * 16 + r + hi * 8;
          float v = lrelu(ab_a[o] * zt[r] + ab_b[o]);
          rm[t][r] = fmaxf(rm[t][r], v);
        }
      }
    }

    // max across the 16 columns held by each lane half (pure VALU butterflies)
#pragma unroll
    for (int t = 0; t < 4; t++)
#pragma unroll
      for (int r = 0; r < 8; r++) {
        float v = rm[t][r];
        v = fmaxf(v, permxor_f<1>(v));
        v = fmaxf(v, permxor_f<2>(v));
        v = fmaxf(v, permxor_f<4>(v));
        v = fmaxf(v, permxor_f<8>(v));
        rm[t][r] = v;
      }

    if (col == 0) {
      int tt = m / NPOINT_, ss = m % NPOINT_;
#pragma unroll
      for (int t = 0; t < 4; t++)
#pragma unroll
        for (int r = 0; r < 8; r++) {
          int o = t * 16 + r + hi * 8;                       // 0..63
          size_t oi = (((size_t)b * 128 + branch * 64 + o) * T_ + tt) * NPOINT_ + ss;
          if (frame == 0) outfeat[oi] = rm[t][r];
          else            outfeat[oi] -= rm[t][r];
        }
    }
  }
}

// ---------------------------------------------------------------------------
extern "C" void kernel_launch(void* const* d_in, const int* in_sizes, int n_in,
                              void* d_out, int out_size, void* d_ws, size_t ws_size,
                              hipStream_t stream) {
  const float* xyz   = (const float*)d_in[0];
  const float* times = (const float*)d_in[1];
  const float* feat  = (const float*)d_in[2];
  const float* W1 = (const float*)d_in[3];
  const float* b1 = (const float*)d_in[4];
  const float* g1 = (const float*)d_in[5];
  const float* be1 = (const float*)d_in[6];
  const float* W2 = (const float*)d_in[7];
  const float* b2 = (const float*)d_in[8];
  const float* g2 = (const float*)d_in[9];
  const float* be2 = (const float*)d_in[10];
  const float* W3 = (const float*)d_in[11];
  const float* b3 = (const float*)d_in[12];
  const float* g3 = (const float*)d_in[13];
  const float* be3 = (const float*)d_in[14];

  float* out = (float*)d_out;
  char* ws = (char*)d_ws;
  // workspace layout (bytes)
  int*   ind     = (int*)(ws + 0);          // 16384 ints
  float* pointsw = (float*)(ws + 65536);    // 49152 f
  float* pnormw  = (float*)(ws + 262144);   // 16384 f
  float* alpha   = (float*)(ws + 327680);   // 256 f
  float* beta    = (float*)(ws + 328704);   // 256 f
  int*   nind    = (int*)(ws + 331776);     // 2 * 1048576 ints
  if (ws_size < (size_t)331776 + 2ull * B_ * M_ * KMAX_ * 4ull) return;

  fps_kernel<<<B_ * T_, 1024, 0, stream>>>(xyz, ind);
  gather_kernel<<<(B_ * T_ * NPOINT_ + 255) / 256, 256, 0, stream>>>(
      xyz, times, ind, out, pointsw, pnormw);
  ab_kernel<<<1, 256, 0, stream>>>(b1, g1, be1, b2, g2, be2, b3, g3, be3, alpha, beta);
  for (int f = 0; f < T_; ++f)
    knn_kernel<<<2048, 256, 0, stream>>>(xyz, pointsw, pnormw,
                                         nind + (size_t)f * B_ * M_ * KMAX_, f);
  float* outfeat = out + (size_t)B_ * T_ * 4 * NPOINT_;   // 65536
  for (int br = 0; br < 2; ++br)
    for (int f = 0; f < T_; ++f)
      mlp_kernel<<<2048, 256, 0, stream>>>(xyz, feat, pointsw,
                                           nind + (size_t)f * B_ * M_ * KMAX_,
                                           W1, W2, W3, alpha, beta, outfeat,
                                           f, br, br == 0 ? 32 : 64);
}
